// TCA_51582557225492
// MI455X (gfx1250) — matrix operand order repaired and checked by reading, not compiled, not run on hardware
//
#include <hip/hip_runtime.h>
#include <math.h>

// ---------------------------------------------------------------------------
// Fused Temporal-Channel Attention for MI455X (gfx1250, wave32).
//
// HBM roofline: x (245.8MB) read once, out (245.8MB) written once (NT stores);
// Phase-B re-read of the per-block 128KB x tile hits L2. MLPs run on
// v_wmma_f32_16x16x32_f16 over 16-voxel M-tiles with zero-padded LDS operands
// so every fragment build is a branch-free ds_load + cvt.
// ---------------------------------------------------------------------------

typedef __attribute__((ext_vector_type(16))) _Float16 v16h;
typedef __attribute__((ext_vector_type(8)))  float    v8f;
typedef __attribute__((ext_vector_type(4)))  float    v4f;

#define TT 10      // num sweeps
#define CC 64      // channels
#define HT 5       // T // RT
#define HC 8       // C // RC
#define PP 32      // points per voxel (== wave32 width)
#define VPB 16     // voxels per block (WMMA M-tile)

// A-fragment K index for v_wmma_*_16x16x32 (16-bit A, ISA 7.12.2):
// lanes 0-15: halves 0..7 -> K0..7, halves 8..15 -> K16..23; lanes 16-31:
// K8..15 / K24..31.  Row M = lane & 15.
__device__ __forceinline__ int ka_map(int lane, int i) {
  return ((i >> 3) << 4) | ((lane >> 4) << 3) | (i & 7);
}
// B-fragment K index (16-bit B, column per lane): same K order as A —
// lanes 0-15: K0..7 / K16..23; lanes 16-31: K8..15 / K24..31.  Col N = lane & 15.
__device__ __forceinline__ int kb_map(int lane, int i) {
  return ((i >> 3) << 4) | ((lane >> 4) << 3) | (i & 7);
}

__device__ __forceinline__ float sigmoidf_fast(float z) {
  return 1.0f / (1.0f + __expf(-z));
}

__global__ __launch_bounds__(256)
void tca_fused_kernel(const float* __restrict__ x,
                      const int*   __restrict__ timestep,
                      const unsigned char* __restrict__ mask,   // jax bool -> 1 byte
                      const float* __restrict__ ta_w1, const float* __restrict__ ta_b1,
                      const float* __restrict__ ta_w2, const float* __restrict__ ta_b2,
                      const float* __restrict__ ca_w1, const float* __restrict__ ca_b1,
                      const float* __restrict__ ca_w2, const float* __restrict__ ca_b2,
                      float* __restrict__ out,
                      int V)
{
  // Reduction results / MLP activations (zero-padded to WMMA K ranges)
  __shared__ float ls_cmax[VPB][CC];   // per-voxel channel max           4 KB
  __shared__ float ls_tmax[VPB][32];   // temporal max, cols 10..31 = 0   2 KB
  __shared__ float ls_hca [VPB][32];   // relu hidden (ca), cols 8..31=0  2 KB
  __shared__ float ls_hta [VPB][32];   // relu hidden (ta), cols 5..31=0  2 KB
  __shared__ float ls_caw [VPB][CC];   // ca_w output                     4 KB
  __shared__ float ls_taw [VPB][16];   // ta_w output                     1 KB
  // Zero-padded weight tiles (B operands) + biases
  __shared__ float lsw_ca1[64][16];    // ca_w1 [64x8] padded N->16       4 KB
  __shared__ float lsw_ca2[32][64];    // ca_w2 [8x64]  padded K->32      8 KB
  __shared__ float lsw_ta1[32][16];    // ta_w1 [10x5]  padded            2 KB
  __shared__ float lsw_ta2[32][16];    // ta_w2 [5x10]  padded            2 KB
  __shared__ float lsb_ca1[16], lsb_ta1[16], lsb_ta2[16], lsb_ca2[64];

  const int tid  = threadIdx.x;
  const int lane = tid & 31;
  const int wv   = tid >> 5;           // wave 0..7
  const int vblk = blockIdx.x * VPB;
  const int mrow = lane & 15;

  // ============ Stage 0: cooperative weight staging (single pass, no race) =
  for (int i = tid; i < 64 * 16; i += 256) {
    const int k = i >> 4, n = i & 15;
    lsw_ca1[k][n] = (n < HC) ? ca_w1[k * HC + n] : 0.0f;
  }
  for (int i = tid; i < 32 * 64; i += 256) {
    const int k = i >> 6, n = i & 63;
    lsw_ca2[k][n] = (k < HC) ? ca_w2[k * CC + n] : 0.0f;
  }
  for (int i = tid; i < 32 * 16; i += 256) {
    const int k = i >> 4, n = i & 15;
    lsw_ta1[k][n] = (k < TT && n < HT) ? ta_w1[k * HT + n] : 0.0f;
    lsw_ta2[k][n] = (k < HT && n < TT) ? ta_w2[k * TT + n] : 0.0f;
  }
  for (int i = tid; i < 64; i += 256) lsb_ca2[i] = ca_b2[i];
  if (tid < 16) {
    lsb_ca1[tid] = (tid < HC) ? ca_b1[tid] : 0.0f;
    lsb_ta1[tid] = (tid < HT) ? ta_b1[tid] : 0.0f;
    lsb_ta2[tid] = (tid < TT) ? ta_b2[tid] : 0.0f;
  }
  // zero the padded activation regions (disjoint from Phase-A writes)
  for (int i = tid; i < VPB * 32; i += 256) {
    const int r = i >> 5, c = i & 31;
    ls_hca[r][c] = 0.0f;
    ls_hta[r][c] = 0.0f;
    if (c >= TT) ls_tmax[r][c] = 0.0f;
  }

  // ============ Phase A: streaming reductions (lane == point index) ========
  for (int vi = 0; vi < 2; ++vi) {
    const int vloc = wv * 2 + vi;
    const int v    = vblk + vloc;
    if (v < V) {
      const float* xp = x + ((size_t)v * PP + lane) * CC;
      float src = -INFINITY;
      #pragma unroll
      for (int c0 = 0; c0 < CC; c0 += 4) {
        const v4f d = *reinterpret_cast<const v4f*>(xp + c0);
        src = fmaxf(src, fmaxf(fmaxf(d.x, d.y), fmaxf(d.z, d.w)));
        float m0 = d.x, m1 = d.y, m2 = d.z, m3 = d.w;
        #pragma unroll
        for (int off = 16; off >= 1; off >>= 1) {   // wave32 tree max
          m0 = fmaxf(m0, __shfl_xor(m0, off));
          m1 = fmaxf(m1, __shfl_xor(m1, off));
          m2 = fmaxf(m2, __shfl_xor(m2, off));
          m3 = fmaxf(m3, __shfl_xor(m3, off));
        }
        if (lane == 0) {
          ls_cmax[vloc][c0 + 0] = m0; ls_cmax[vloc][c0 + 1] = m1;
          ls_cmax[vloc][c0 + 2] = m2; ls_cmax[vloc][c0 + 3] = m3;
        }
      }
      const int  ts = timestep[(size_t)v * PP + lane];
      const bool mk = mask[(size_t)v * PP + lane] != 0;
      #pragma unroll
      for (int t = 0; t < TT; ++t) {
        // max over points of where(sel, src, 0)  (matches jnp.where -> max)
        float val = (mk && ts == t) ? src : 0.0f;
        #pragma unroll
        for (int off = 16; off >= 1; off >>= 1)
          val = fmaxf(val, __shfl_xor(val, off));
        if (lane == 0) ls_tmax[vloc][t] = val;
      }
    } else if (lane < 16) {
      // pad rows so WMMA A-fragments read defined values
      #pragma unroll
      for (int c0 = lane; c0 < CC; c0 += 16) ls_cmax[vloc][c0] = 0.0f;
      if (lane < TT) ls_tmax[vloc][lane] = 0.0f;
    }
  }
  __syncthreads();

  // ============ Stage 1: hidden layers via WMMA (branch-free frag builds) ==
  if (wv == 0) {
    // channel attention L1: [16 x 64] @ [64 x 8]  (K = 2 chunks of 32)
    v8f acc = {};
    #pragma unroll
    for (int kc = 0; kc < 2; ++kc) {
      v16h a, b;
      #pragma unroll
      for (int i = 0; i < 16; ++i) {
        a[i] = (_Float16)ls_cmax[mrow][kc * 32 + ka_map(lane, i)];
        b[i] = (_Float16)lsw_ca1[kc * 32 + kb_map(lane, i)][mrow];
      }
      acc = __builtin_amdgcn_wmma_f32_16x16x32_f16(false, a, false, b,
                                                   (short)0, acc, false, false);
    }
    if (mrow < HC) {
      #pragma unroll
      for (int r = 0; r < 8; ++r) {
        const int M = r + ((lane >> 4) << 3);
        ls_hca[M][mrow] = fmaxf(acc[r] + lsb_ca1[mrow], 0.0f);
      }
    }
  } else if (wv == 1) {
    // temporal attention L1: [16 x 10] @ [10 x 5]  (zero-padded operands)
    v16h a, b;
    #pragma unroll
    for (int i = 0; i < 16; ++i) {
      a[i] = (_Float16)ls_tmax[mrow][ka_map(lane, i)];
      b[i] = (_Float16)lsw_ta1[kb_map(lane, i)][mrow];
    }
    v8f acc = {};
    acc = __builtin_amdgcn_wmma_f32_16x16x32_f16(false, a, false, b,
                                                 (short)0, acc, false, false);
    if (mrow < HT) {
      #pragma unroll
      for (int r = 0; r < 8; ++r) {
        const int M = r + ((lane >> 4) << 3);
        ls_hta[M][mrow] = fmaxf(acc[r] + lsb_ta1[mrow], 0.0f);
      }
    }
  }
  __syncthreads();

  // ============ Stage 2: output layers via WMMA ============================
  if (wv == 0) {
    // channel attention L2: [16 x 8] @ [8 x 64]  (4 N-tiles of 16)
    v16h a2;
    #pragma unroll
    for (int i = 0; i < 16; ++i)
      a2[i] = (_Float16)ls_hca[mrow][ka_map(lane, i)];
    #pragma unroll
    for (int nt = 0; nt < 4; ++nt) {
      v16h b2;
      #pragma unroll
      for (int i = 0; i < 16; ++i)
        b2[i] = (_Float16)lsw_ca2[kb_map(lane, i)][nt * 16 + mrow];
      v8f c2 = {};
      c2 = __builtin_amdgcn_wmma_f32_16x16x32_f16(false, a2, false, b2,
                                                  (short)0, c2, false, false);
      #pragma unroll
      for (int r = 0; r < 8; ++r) {
        const int M = r + ((lane >> 4) << 3);
        ls_caw[M][nt * 16 + mrow] = c2[r] + lsb_ca2[nt * 16 + mrow];
      }
    }
  } else if (wv == 1) {
    // temporal attention L2: [16 x 5] @ [5 x 10]
    v16h a2, b2;
    #pragma unroll
    for (int i = 0; i < 16; ++i) {
      a2[i] = (_Float16)ls_hta[mrow][ka_map(lane, i)];
      b2[i] = (_Float16)lsw_ta2[kb_map(lane, i)][mrow];
    }
    v8f c2 = {};
    c2 = __builtin_amdgcn_wmma_f32_16x16x32_f16(false, a2, false, b2,
                                                (short)0, c2, false, false);
    if (mrow < TT) {
      #pragma unroll
      for (int r = 0; r < 8; ++r) {
        const int M = r + ((lane >> 4) << 3);
        ls_taw[M][mrow] = c2[r] + lsb_ta2[mrow];
      }
    }
  }
  __syncthreads();

  // ============ Phase B: apply (x tile is L2-hot; NT last-use/streaming) ===
  for (int vi = 0; vi < 2; ++vi) {
    const int vloc = wv * 2 + vi;
    const int v    = vblk + vloc;
    if (v >= V) continue;
    const size_t base = ((size_t)v * PP + lane) * CC;
    const int  ts = timestep[(size_t)v * PP + lane];
    const bool mk = mask[(size_t)v * PP + lane] != 0;
    const float tav = ls_taw[vloc][ts];
    #pragma unroll
    for (int c0 = 0; c0 < CC; c0 += 4) {
      const v4f d = __builtin_nontemporal_load(
          reinterpret_cast<const v4f*>(x + base + c0));
      v4f o;
      o.x = mk ? d.x * sigmoidf_fast(tav * ls_caw[vloc][c0 + 0]) : 0.0f;
      o.y = mk ? d.y * sigmoidf_fast(tav * ls_caw[vloc][c0 + 1]) : 0.0f;
      o.z = mk ? d.z * sigmoidf_fast(tav * ls_caw[vloc][c0 + 2]) : 0.0f;
      o.w = mk ? d.w * sigmoidf_fast(tav * ls_caw[vloc][c0 + 3]) : 0.0f;
      __builtin_nontemporal_store(o, reinterpret_cast<v4f*>(out + base + c0));
    }
  }
}

extern "C" void kernel_launch(void* const* d_in, const int* in_sizes, int n_in,
                              void* d_out, int out_size, void* d_ws, size_t ws_size,
                              hipStream_t stream) {
  const float*         x     = (const float*)d_in[0];
  const int*           ts    = (const int*)d_in[1];
  const unsigned char* mk    = (const unsigned char*)d_in[2];   // jax bool = 1 byte
  const float*         ta_w1 = (const float*)d_in[3];
  const float*         ta_b1 = (const float*)d_in[4];
  const float*         ta_w2 = (const float*)d_in[5];
  const float*         ta_b2 = (const float*)d_in[6];
  const float*         ca_w1 = (const float*)d_in[7];
  const float*         ca_b1 = (const float*)d_in[8];
  const float*         ca_w2 = (const float*)d_in[9];
  const float*         ca_b2 = (const float*)d_in[10];
  float*               out   = (float*)d_out;

  const int V = in_sizes[1] / PP;          // timestep is [V, P]
  const int nblocks = (V + VPB - 1) / VPB; // 1875 for V=30000

  tca_fused_kernel<<<nblocks, 256, 0, stream>>>(
      x, ts, mk, ta_w1, ta_b1, ta_w2, ta_b2,
      ca_w1, ca_b1, ca_w2, ca_b2, out, V);
}
